// TokenPacker_75625784148330
// MI455X (gfx1250) — compile-verified, matrix operand order checked by reading, not run on hardware
//
#include <hip/hip_runtime.h>

typedef unsigned short u16;
typedef unsigned int   u32;
typedef unsigned long long u64;

typedef __attribute__((ext_vector_type(16))) __bf16 v16bf;
typedef __attribute__((ext_vector_type(8)))  __bf16 v8bf;
typedef __attribute__((ext_vector_type(8)))  float  v8f;

#define E_DIM 1024

// ---------- helpers ----------
__device__ __forceinline__ u16 f2bf(float x) {
  u32 u = __float_as_uint(x);
  u32 r = u + 0x7FFFu + ((u >> 16) & 1u);   // round-to-nearest-even
  return (u16)(r >> 16);
}
__device__ __forceinline__ float bflo(u32 w) { return __uint_as_float(w << 16); }
__device__ __forceinline__ float bfhi(u32 w) { return __uint_as_float(w & 0xffff0000u); }

__device__ __forceinline__ float wave_xor_add(float x, int m) {
  int lane = (int)(threadIdx.x & 31);
  int v = __builtin_amdgcn_ds_bpermute(((lane ^ m) << 2), __float_as_int(x));
  return x + __int_as_float(v);
}

// CDNA5 async global->LDS copy (ASYNCcnt-tracked, no VGPR staging).
// ldsOff = LDS byte offset (low 32 bits of generic shared address), g = per-lane global src.
__device__ __forceinline__ void async_ld_b128(u32 ldsOff, const void* g) {
  asm volatile("global_load_async_to_lds_b128 %0, %1, off"
               :: "v"(ldsOff), "v"((u64)(uintptr_t)g)
               : "memory");
}
__device__ __forceinline__ void wait_async0() {
  asm volatile("s_wait_asynccnt 0" ::: "memory");
}

// ---------- fp32 -> bf16 convert ----------
__global__ void k_cvt_bf16(const float4* __restrict__ in, uint2* __restrict__ out, long n4) {
  long i = (long)blockIdx.x * blockDim.x + threadIdx.x;
  long stride = (long)gridDim.x * blockDim.x;
  for (; i < n4; i += stride) {
    float4 v = in[i];
    uint2 o;
    o.x = (u32)f2bf(v.x) | ((u32)f2bf(v.y) << 16);
    o.y = (u32)f2bf(v.z) | ((u32)f2bf(v.w) << 16);
    out[i] = o;
  }
}

// ---------- WMMA GEMM: C[M x 1024] = A[M x 1024](bf16) * B[1024 x 1024](bf16, row=out col)^T + bias ----------
// EPI: 0 = fp32 out, 1 = bf16 out, 2 = exact-GELU then bf16 out
template<int EPI>
__global__ void __launch_bounds__(256)
k_gemm_bf16(const u16* __restrict__ A, const u16* __restrict__ B,
            const float* __restrict__ bias, void* __restrict__ Cout, int M)
{
  // double-buffered LDS: per buf, A tile 128 rows x 80B (32 bf16 + 16B pad), B tile same
  __shared__ char smem[2 * 20480];
  const int tid  = threadIdx.x;
  const int lane = tid & 31;
  const int wave = tid >> 5;      // 8 waves
  const int wm   = wave >> 1;     // 0..3 : 32-row slab
  const int wn   = wave & 1;      // 0..1 : 64-col slab
  const int tileM = blockIdx.x * 128;
  const int tileN = blockIdx.y * 128;
  const u32 smemBase = (u32)(uintptr_t)(&smem[0]);   // LDS byte offset of tile base

  v8f acc[2][4];
#pragma unroll
  for (int i = 0; i < 2; ++i)
#pragma unroll
    for (int j = 0; j < 4; ++j)
#pragma unroll
      for (int r = 0; r < 8; ++r) acc[i][j][r] = 0.0f;

  // Issue async global->LDS copies for one 128x32 A tile + 128x32 B tile.
  auto stage_load = [&](int kk, int buf) {
    const u32 aBase = smemBase + buf * 20480;
    const u32 bBase = aBase + 10240;
#pragma unroll
    for (int i = 0; i < 2; ++i) {
      int c   = tid + i * 256;      // 512 x 16B chunks per matrix
      int row = c >> 2;
      int seg = c & 3;
      const u16* ag = A + (size_t)(tileM + row) * E_DIM + kk + seg * 8;
      const u16* bg = B + (size_t)(tileN + row) * E_DIM + kk + seg * 8;
      if (kk + 64 < E_DIM) {        // prefetch 2 stages ahead -> global_prefetch_b8
        __builtin_prefetch(ag + 64, 0, 1);
        __builtin_prefetch(bg + 64, 0, 1);
      }
      async_ld_b128(aBase + row * 80 + seg * 16, ag);
      async_ld_b128(bBase + row * 80 + seg * 16, bg);
    }
  };

  // load one 16x32 bf16 fragment from an LDS tile (rows of 32 bf16, stride 80B).
  // ISA layout: lanes 0-15 hold K 0..7 & 16..23, lanes 16-31 hold K 8..15 & 24..31.
  auto frag = [&](const char* base, int fr) -> v16bf {
    int r = fr * 16 + (lane & 15);
    const char* p = base + r * 80 + ((lane >> 4) & 1) * 16;
    union { v16bf v; v8bf h[2]; } u;
    u.h[0] = *(const v8bf*)p;
    u.h[1] = *(const v8bf*)(p + 32);
    return u.v;
  };

  stage_load(0, 0);
  wait_async0();
  __syncthreads();

  for (int ks = 0; ks < E_DIM / 32; ++ks) {
    int buf = ks & 1;
    if (ks + 1 < E_DIM / 32) stage_load((ks + 1) * 32, buf ^ 1);
    const char* aL = smem + buf * 20480;
    const char* bL = aL + 10240;
    v16bf af[2], bfr[4];
#pragma unroll
    for (int fm = 0; fm < 2; ++fm) af[fm] = frag(aL, wm * 2 + fm);
#pragma unroll
    for (int fn = 0; fn < 4; ++fn) bfr[fn] = frag(bL, wn * 4 + fn);
#pragma unroll
    for (int fm = 0; fm < 2; ++fm)
#pragma unroll
      for (int fn = 0; fn < 4; ++fn)
        acc[fm][fn] = __builtin_amdgcn_wmma_f32_16x16x32_bf16(
            false, af[fm], false, bfr[fn], (short)0, acc[fm][fn], false, false);
    wait_async0();       // next-stage tiles resident in LDS
    __syncthreads();     // all waves done reading current buffer & filling next
  }

  // epilogue: C frag layout -> col = lane&15, rows = fragRow + (lane>=16?8:0) + vgpr
#pragma unroll
  for (int fm = 0; fm < 2; ++fm)
#pragma unroll
    for (int fn = 0; fn < 4; ++fn) {
      int col  = tileN + wn * 64 + fn * 16 + (lane & 15);
      int row0 = tileM + wm * 32 + fm * 16 + ((lane >> 4) & 1) * 8;
      float bv = bias ? bias[col] : 0.0f;
#pragma unroll
      for (int r = 0; r < 8; ++r) {
        float x = acc[fm][fn][r] + bv;
        if (EPI == 2) x = 0.5f * x * (1.0f + erff(x * 0.70710678118654752f));
        size_t o = (size_t)(row0 + r) * E_DIM + col;
        if (EPI == 0) ((float*)Cout)[o] = x;
        else          ((u16*)Cout)[o]   = f2bf(x);
      }
    }
}

// ---------- LayerNorm(1024) fp32 -> bf16, one wave32 per row ----------
__global__ void __launch_bounds__(256)
k_layernorm(const float* __restrict__ in, const float* __restrict__ gw,
            const float* __restrict__ gb, u16* __restrict__ out, int rows)
{
  int lane = threadIdx.x & 31;
  int wave = threadIdx.x >> 5;
  int row  = blockIdx.x * 8 + wave;
  if (row >= rows) return;
  const float4* p = (const float4*)(in + (size_t)row * E_DIM);
  float4 v[8];
  float s = 0.f, s2 = 0.f;
#pragma unroll
  for (int j = 0; j < 8; ++j) {
    v[j] = p[j * 32 + lane];
    s  += v[j].x + v[j].y + v[j].z + v[j].w;
    s2 += v[j].x * v[j].x + v[j].y * v[j].y + v[j].z * v[j].z + v[j].w * v[j].w;
  }
#pragma unroll
  for (int m = 16; m >= 1; m >>= 1) { s = wave_xor_add(s, m); s2 = wave_xor_add(s2, m); }
  float mean = s * (1.0f / E_DIM);
  float var  = s2 * (1.0f / E_DIM) - mean * mean;
  float rstd = rsqrtf(var + 1e-6f);
  const float4* wp = (const float4*)gw;
  const float4* bp = (const float4*)gb;
  uint2* op = (uint2*)(out + (size_t)row * E_DIM);
#pragma unroll
  for (int j = 0; j < 8; ++j) {
    float4 w4 = wp[j * 32 + lane];
    float4 b4 = bp[j * 32 + lane];
    float o0 = (v[j].x - mean) * rstd * w4.x + b4.x;
    float o1 = (v[j].y - mean) * rstd * w4.y + b4.y;
    float o2 = (v[j].z - mean) * rstd * w4.z + b4.z;
    float o3 = (v[j].w - mean) * rstd * w4.w + b4.w;
    uint2 o;
    o.x = (u32)f2bf(o0) | ((u32)f2bf(o1) << 16);
    o.y = (u32)f2bf(o2) | ((u32)f2bf(o3) << 16);
    op[j * 32 + lane] = o;
  }
}

// ---------- attention: 1 query x 4 keys per (token, head); thread = (b, h) ----------
// Qh rows: b = n*144 + q ; Kh/Vh rows: n*576 + t ; O rows: n*144 + q (output order)
__global__ void __launch_bounds__(256)
k_attn(const u16* __restrict__ Qh, const u16* __restrict__ Kh,
       const u16* __restrict__ Vh, u16* __restrict__ O, int total)
{
  int gid = blockIdx.x * blockDim.x + threadIdx.x;
  if (gid >= total) return;
  int h = gid & 7;
  int b = gid >> 3;
  int n = b / 144;
  int q = b % 144;
  int a = q / 12, c = q % 12;

  size_t rowKV[4];
#pragma unroll
  for (int kk = 0; kk < 4; ++kk) {
    int t = (2 * a + (kk >> 1)) * 24 + (2 * c + (kk & 1));   // _divide(S=2) gather
    rowKV[kk] = (size_t)(n * 576 + t);
  }

  const uint4* qp = (const uint4*)(Qh + (size_t)b * E_DIM + h * 128);
  uint4 qv[16];
#pragma unroll
  for (int j = 0; j < 16; ++j) qv[j] = qp[j];

  float sc[4];
#pragma unroll
  for (int kk = 0; kk < 4; ++kk) {
    const uint4* kp = (const uint4*)(Kh + rowKV[kk] * E_DIM + h * 128);
    float d = 0.f;
#pragma unroll
    for (int j = 0; j < 16; ++j) {
      uint4 kv = kp[j];
      d += bflo(qv[j].x) * bflo(kv.x) + bfhi(qv[j].x) * bfhi(kv.x);
      d += bflo(qv[j].y) * bflo(kv.y) + bfhi(qv[j].y) * bfhi(kv.y);
      d += bflo(qv[j].z) * bflo(kv.z) + bfhi(qv[j].z) * bfhi(kv.z);
      d += bflo(qv[j].w) * bflo(kv.w) + bfhi(qv[j].w) * bfhi(kv.w);
    }
    sc[kk] = d * 0.08838834764831845f;   // 1/sqrt(128)
  }

  float mx = fmaxf(fmaxf(sc[0], sc[1]), fmaxf(sc[2], sc[3]));
  float pe[4], ps = 0.f;
#pragma unroll
  for (int kk = 0; kk < 4; ++kk) { pe[kk] = __expf(sc[kk] - mx); ps += pe[kk]; }
  float inv = 1.0f / ps;
#pragma unroll
  for (int kk = 0; kk < 4; ++kk) pe[kk] *= inv;

  const uint4* v0 = (const uint4*)(Vh + rowKV[0] * E_DIM + h * 128);
  const uint4* v1 = (const uint4*)(Vh + rowKV[1] * E_DIM + h * 128);
  const uint4* v2 = (const uint4*)(Vh + rowKV[2] * E_DIM + h * 128);
  const uint4* v3 = (const uint4*)(Vh + rowKV[3] * E_DIM + h * 128);
  uint4* op = (uint4*)(O + (size_t)b * E_DIM + h * 128);
#pragma unroll
  for (int j = 0; j < 16; ++j) {
    uint4 a0 = v0[j], a1 = v1[j], a2 = v2[j], a3 = v3[j];
    float o0 = pe[0]*bflo(a0.x) + pe[1]*bflo(a1.x) + pe[2]*bflo(a2.x) + pe[3]*bflo(a3.x);
    float o1 = pe[0]*bfhi(a0.x) + pe[1]*bfhi(a1.x) + pe[2]*bfhi(a2.x) + pe[3]*bfhi(a3.x);
    float o2 = pe[0]*bflo(a0.y) + pe[1]*bflo(a1.y) + pe[2]*bflo(a2.y) + pe[3]*bflo(a3.y);
    float o3 = pe[0]*bfhi(a0.y) + pe[1]*bfhi(a1.y) + pe[2]*bfhi(a2.y) + pe[3]*bfhi(a3.y);
    float o4 = pe[0]*bflo(a0.z) + pe[1]*bflo(a1.z) + pe[2]*bflo(a2.z) + pe[3]*bflo(a3.z);
    float o5 = pe[0]*bfhi(a0.z) + pe[1]*bfhi(a1.z) + pe[2]*bfhi(a2.z) + pe[3]*bfhi(a3.z);
    float o6 = pe[0]*bflo(a0.w) + pe[1]*bflo(a1.w) + pe[2]*bflo(a2.w) + pe[3]*bflo(a3.w);
    float o7 = pe[0]*bfhi(a0.w) + pe[1]*bfhi(a1.w) + pe[2]*bfhi(a2.w) + pe[3]*bfhi(a3.w);
    uint4 o;
    o.x = (u32)f2bf(o0) | ((u32)f2bf(o1) << 16);
    o.y = (u32)f2bf(o2) | ((u32)f2bf(o3) << 16);
    o.z = (u32)f2bf(o4) | ((u32)f2bf(o5) << 16);
    o.w = (u32)f2bf(o6) | ((u32)f2bf(o7) << 16);
    op[j] = o;
  }
}

// ---------- host orchestration ----------
extern "C" void kernel_launch(void* const* d_in, const int* in_sizes, int n_in,
                              void* d_out, int out_size, void* d_ws, size_t ws_size,
                              hipStream_t stream)
{
  (void)in_sizes; (void)n_in; (void)out_size; (void)ws_size;
  const float* x      = (const float*)d_in[0];
  const float* x_feat = (const float*)d_in[1];
  const float* w_q    = (const float*)d_in[2];
  const float* w_k1   = (const float*)d_in[3];
  const float* b_k1   = (const float*)d_in[4];
  const float* w_k2   = (const float*)d_in[5];
  const float* b_k2   = (const float*)d_in[6];
  const float* w_v1   = (const float*)d_in[7];
  const float* b_v1   = (const float*)d_in[8];
  const float* w_v2   = (const float*)d_in[9];
  const float* b_v2   = (const float*)d_in[10];
  const float* ln_q_w = (const float*)d_in[11];
  const float* ln_q_b = (const float*)d_in[12];
  const float* ln_k_w = (const float*)d_in[13];
  const float* ln_k_b = (const float*)d_in[14];
  const float* ln_v_w = (const float*)d_in[15];
  const float* ln_v_b = (const float*)d_in[16];
  const float* in_w   = (const float*)d_in[17];
  const float* in_b   = (const float*)d_in[18];
  const float* out_w  = (const float*)d_in[19];
  const float* out_b  = (const float*)d_in[20];

  const size_t E  = 1024;
  const size_t MK = 36864;   // 64 * 576  (key/value tokens)
  const size_t MQ = 9216;    // 64 * 144  (query tokens)

  char* ws = (char*)d_ws;
  size_t cur = 0;
  auto take = [&](size_t bytes) -> char* {
    char* p = ws + cur;
    cur += (bytes + 255) & ~(size_t)255;
    return p;
  };
  u16*  xf_bf = (u16*)take(MK * E * 2);
  u16*  x_bf  = (u16*)take(MQ * E * 2);
  u16*  wk1b  = (u16*)take(E * E * 2);
  u16*  wk2b  = (u16*)take(E * E * 2);
  u16*  wv1b  = (u16*)take(E * E * 2);
  u16*  wv2b  = (u16*)take(E * E * 2);
  u16*  wqb   = (u16*)take(E * E * 2);
  u16*  inwb  = (u16*)take(3 * E * E * 2);
  u16*  outwb = (u16*)take(E * E * 2);
  u16*  t_bf  = (u16*)take(MK * E * 2);
  float* pre  = (float*)take(MK * E * 4);
  u16*  klnb  = (u16*)take(MK * E * 2);
  u16*  vlnb  = (u16*)take(MK * E * 2);
  u16*  qlnb  = (u16*)take(MQ * E * 2);
  u16*  Khb   = (u16*)take(MK * E * 2);
  u16*  Vhb   = (u16*)take(MK * E * 2);
  u16*  Qhb   = (u16*)take(MQ * E * 2);
  u16*  obf   = (u16*)take(MQ * E * 2);

  auto cvt = [&](const float* src, u16* dst, size_t n) {
    k_cvt_bf16<<<1024, 256, 0, stream>>>((const float4*)src, (uint2*)dst, (long)(n / 4));
  };
  cvt(x, x_bf, MQ * E);
  cvt(x_feat, xf_bf, MK * E);
  cvt(w_q, wqb, E * E);
  cvt(w_k1, wk1b, E * E);
  cvt(w_k2, wk2b, E * E);
  cvt(w_v1, wv1b, E * E);
  cvt(w_v2, wv2b, E * E);
  cvt(in_w, inwb, 3 * E * E);
  cvt(out_w, outwb, E * E);

  dim3 blk(256);
  dim3 gKV((unsigned)(MK / 128), 8);
  dim3 gQ((unsigned)(MQ / 128), 8);

  // key path: gelu(x_feat @ w_k1^T + b) -> @ w_k2^T + b -> LN
  k_gemm_bf16<2><<<gKV, blk, 0, stream>>>(xf_bf, wk1b, b_k1, t_bf, (int)MK);
  k_gemm_bf16<0><<<gKV, blk, 0, stream>>>(t_bf, wk2b, b_k2, pre, (int)MK);
  k_layernorm<<<(unsigned)(MK / 8), blk, 0, stream>>>(pre, ln_k_w, ln_k_b, klnb, (int)MK);
  // value path
  k_gemm_bf16<2><<<gKV, blk, 0, stream>>>(xf_bf, wv1b, b_v1, t_bf, (int)MK);
  k_gemm_bf16<0><<<gKV, blk, 0, stream>>>(t_bf, wv2b, b_v2, pre, (int)MK);
  k_layernorm<<<(unsigned)(MK / 8), blk, 0, stream>>>(pre, ln_v_w, ln_v_b, vlnb, (int)MK);
  // query path (no bias on q_proj_1)
  k_gemm_bf16<0><<<gQ, blk, 0, stream>>>(x_bf, wqb, nullptr, pre, (int)MQ);
  k_layernorm<<<(unsigned)(MQ / 8), blk, 0, stream>>>(pre, ln_q_w, ln_q_b, qlnb, (int)MQ);
  // MHA in-projections (in_w split into q/k/v blocks of 1024 rows each)
  k_gemm_bf16<1><<<gQ,  blk, 0, stream>>>(qlnb, inwb,             in_b,           Qhb, (int)MQ);
  k_gemm_bf16<1><<<gKV, blk, 0, stream>>>(klnb, inwb + E * E,     in_b + E,       Khb, (int)MK);
  k_gemm_bf16<1><<<gKV, blk, 0, stream>>>(vlnb, inwb + 2 * E * E, in_b + 2 * E,   Vhb, (int)MK);
  // attention (1 query x 4 pooled keys per head), writes o in final row order
  int total = (int)(MQ * 8);
  k_attn<<<(total + 255) / 256, blk, 0, stream>>>(Qhb, Khb, Vhb, obf, total);
  // out projection straight into d_out (fp32)
  k_gemm_bf16<0><<<gQ, blk, 0, stream>>>(obf, outwb, out_b, (float*)d_out, (int)MQ);
}
